// SOM_31610959298600
// MI455X (gfx1250) — matrix operand order repaired and checked by reading, not compiled, not run on hardware
//
#include <hip/hip_runtime.h>

typedef __attribute__((ext_vector_type(16))) __bf16 v16bf;
typedef __attribute__((ext_vector_type(8)))  __bf16 v8bf;
typedef __attribute__((ext_vector_type(8)))  float  v8f;

#define BROWS 2048      // B
#define DK    1024      // D (K dim)
#define MCOLS 16384     // H*W (N dim)
#define EPSV  1e-6f
#define TN    128       // cols per block (8 waves x 16)
#define KT    32        // K chunk per WMMA step
#define LDT   40        // ushort stride (32 + 8 pad) -> 80B, 16B-aligned, conflict-free b128

// ---- workspace layout (bytes) ----
#define OFF_GMIN    0ull                         // 2048 * 8
#define OFF_ROWSUM  16384ull                     // 2048 * 4
#define OFF_ROWSQ   24576ull                     // 2048 * 4
#define OFF_COLSUM  32768ull                     // 16384 * 4
#define OFF_COLSQ   98304ull                     // 16384 * 4
#define OFF_WHI     262144ull                    // 16384*1024*2 = 32 MB (transposed [M][K])
#define OFF_WLO     (OFF_WHI + 33554432ull)
#define OFF_XHI     (OFF_WLO + 33554432ull)      // 2048*1024*2 = 4 MB  ([B][K])
#define OFF_XLO     (OFF_XHI + 4194304ull)
#define WS_NEEDED   (OFF_XLO + 4194304ull)       // ~72.3 MB

__device__ __forceinline__ unsigned short f2bf_rne(float f) {
    unsigned int u = __float_as_uint(f);
    u += 0x7FFFu + ((u >> 16) & 1u);
    return (unsigned short)(u >> 16);
}

// generic pointer into LDS aperture: low 32 bits == wave-relative LDS byte offset
__device__ __forceinline__ unsigned lds_addr_of(const void* p) {
    return (unsigned)(unsigned long long)p;
}

// ASYNCcnt-tracked direct memory->LDS copy (no VGPR round trip)
__device__ __forceinline__ void async_b128(unsigned ldsoff, const unsigned short* g) {
#if defined(__AMDGCN__)
    asm volatile("global_load_async_to_lds_b128 %0, %1, off"
                 :: "v"(ldsoff), "v"(g) : "memory");
#endif
}
__device__ __forceinline__ void wait_async0() {
#if defined(__AMDGCN__)
    asm volatile("s_wait_asynccnt 0x0" ::: "memory");
#endif
}

__global__ __launch_bounds__(256)
void som_init_kernel(unsigned long long* __restrict__ gmin, float* __restrict__ out) {
    int i = blockIdx.x * blockDim.x + threadIdx.x;
    if (i < BROWS) gmin[i] = ~0ull;
    if (i == 0) out[3 * BROWS] = 0.0f;   // loss accumulator slot
}

// ---- pre-pass: weight f32 [K][M] -> bf16 hi/lo planes transposed [M][K], + column stats ----
__global__ __launch_bounds__(256)
void som_prep_w(const float* __restrict__ w, unsigned short* __restrict__ whiT,
                unsigned short* __restrict__ wloT, float* __restrict__ colSum,
                float* __restrict__ colSq) {
    const int m = blockIdx.x * 256 + threadIdx.x;   // one column per thread (coalesced reads)
    float s = 0.f, q = 0.f;
    for (int k = 0; k < DK; k += 8) {
        alignas(16) unsigned short hs[8], ls[8];
        #pragma unroll
        for (int j = 0; j < 8; ++j) {
            float v = w[(size_t)(k + j) * MCOLS + m];
            s += v; q += v * v;
            unsigned short h = f2bf_rne(v);
            float hf = __uint_as_float((unsigned int)h << 16);
            hs[j] = h; ls[j] = f2bf_rne(v - hf);
        }
        *(uint4*)&whiT[(size_t)m * DK + k] = *(const uint4*)hs;  // line-local: L2 write-combines
        *(uint4*)&wloT[(size_t)m * DK + k] = *(const uint4*)ls;
    }
    colSum[m] = s; colSq[m] = q;
}

// ---- pre-pass: x f32 [B][K] -> bf16 hi/lo planes [B][K], + row stats (one block per row) ----
__global__ __launch_bounds__(256)
void som_prep_x(const float* __restrict__ x, unsigned short* __restrict__ xhi,
                unsigned short* __restrict__ xlo, float* __restrict__ rowSum,
                float* __restrict__ rowSq) {
    const int b = blockIdx.x;
    const int t = threadIdx.x;
    const float4 v = *(const float4*)(x + (size_t)b * DK + 4 * t);
    float e[4] = {v.x, v.y, v.z, v.w};
    alignas(8) unsigned short hs[4], ls[4];
    float s = 0.f, q = 0.f;
    #pragma unroll
    for (int j = 0; j < 4; ++j) {
        s += e[j]; q += e[j] * e[j];
        unsigned short h = f2bf_rne(e[j]);
        float hf = __uint_as_float((unsigned int)h << 16);
        hs[j] = h; ls[j] = f2bf_rne(e[j] - hf);
    }
    *(uint2*)&xhi[(size_t)b * DK + 4 * t] = *(const uint2*)hs;
    *(uint2*)&xlo[(size_t)b * DK + 4 * t] = *(const uint2*)ls;
    #pragma unroll
    for (int off = 16; off > 0; off >>= 1) {
        s += __shfl_xor(s, off, 32);
        q += __shfl_xor(q, off, 32);
    }
    __shared__ float sw[8], sq[8];
    if ((t & 31) == 0) { sw[t >> 5] = s; sq[t >> 5] = q; }
    __syncthreads();
    if (t == 0) {
        float S = 0.f, Q = 0.f;
        #pragma unroll
        for (int i = 0; i < 8; ++i) { S += sw[i]; Q += sq[i]; }
        rowSum[b] = S; rowSq[b] = Q;
    }
}

// ---- main GEMM: 32 rows x 128 cols per block; double-buffered async LDS; 6 WMMAs/chunk ----
__global__ __launch_bounds__(256)
void som_gemm_kernel(const unsigned short* __restrict__ xhi, const unsigned short* __restrict__ xlo,
                     const unsigned short* __restrict__ whiT, const unsigned short* __restrict__ wloT,
                     const float* __restrict__ rowSum, const float* __restrict__ rowSq,
                     const float* __restrict__ colSum, const float* __restrict__ colSq,
                     unsigned long long* __restrict__ gmin) {
    __shared__ alignas(16) unsigned short AhiB[2][32 * LDT];
    __shared__ alignas(16) unsigned short AloB[2][32 * LDT];
    __shared__ alignas(16) unsigned short BhiB[2][TN * LDT];
    __shared__ alignas(16) unsigned short BloB[2][TN * LDT];
    __shared__ float rowS[32], rowQ[32];
    __shared__ unsigned long long rowMin[32];

    const int tid = threadIdx.x;
    const int m0  = blockIdx.x * 32;
    const int n0  = blockIdx.y * TN;

    if (tid < 32) { rowS[tid] = rowSum[m0 + tid]; rowQ[tid] = rowSq[m0 + tid]; rowMin[tid] = ~0ull; }

    // A staging: 256 b128 chunks (2 planes x 32 rows x 4) -> one async load per thread
    {
    }
    const unsigned short* aSrcP = (tid & 128) ? xlo : xhi;
    unsigned short (*Ad)[32 * LDT] = (tid & 128) ? AloB : AhiB;
    const int aRow = (tid & 127) >> 2;
    const int aK8  = (tid & 3) * 8;
    const unsigned short* aG = aSrcP + (size_t)(m0 + aRow) * DK + aK8;
    unsigned aL[2];
    aL[0] = lds_addr_of(&Ad[0][aRow * LDT + aK8]);
    aL[1] = lds_addr_of(&Ad[1][aRow * LDT + aK8]);

    // B staging: 1024 b128 chunks (2 planes x 128 cols x 4) -> 4 async loads per thread
    const unsigned short* bG[4];
    unsigned bL[4][2];
    #pragma unroll
    for (int i = 0; i < 4; ++i) {
        const int id  = tid + 256 * i;
        const unsigned short* src = (id & 512) ? wloT : whiT;
        unsigned short (*Bd)[TN * LDT] = (id & 512) ? BloB : BhiB;
        const int r   = id & 511;
        const int col = r >> 2;
        const int k8  = (r & 3) * 8;
        bG[i]    = src + (size_t)(n0 + col) * DK + k8;
        bL[i][0] = lds_addr_of(&Bd[0][col * LDT + k8]);
        bL[i][1] = lds_addr_of(&Bd[1][col * LDT + k8]);
    }

    const int lane = tid & 31;
    const int wave = tid >> 5;
    const int mrow = lane & 15;   // A row / B column within 16x16 tile
    const int half = lane >> 4;   // lane-group per ISA layout
    const int nb   = wave * 16 + mrow;

    v8f acc0 = {}, acc1 = {};

    auto issue = [&](int kc, int buf) {
        async_b128(aL[buf], aG + kc);
        #pragma unroll
        for (int i = 0; i < 4; ++i) async_b128(bL[i][buf], bG[i] + kc);
    };

    auto compute = [&](int buf) {
        const unsigned short* Ah = AhiB[buf];
        const unsigned short* Al = AloB[buf];
        const unsigned short* Bh = BhiB[buf];
        const unsigned short* Bl = BloB[buf];
        v8bf a0h0 = *(const v8bf*)&Ah[mrow * LDT +      half * 8];
        v8bf a0h1 = *(const v8bf*)&Ah[mrow * LDT + 16 + half * 8];
        v8bf a0l0 = *(const v8bf*)&Al[mrow * LDT +      half * 8];
        v8bf a0l1 = *(const v8bf*)&Al[mrow * LDT + 16 + half * 8];
        v8bf a1h0 = *(const v8bf*)&Ah[(mrow + 16) * LDT +      half * 8];
        v8bf a1h1 = *(const v8bf*)&Ah[(mrow + 16) * LDT + 16 + half * 8];
        v8bf a1l0 = *(const v8bf*)&Al[(mrow + 16) * LDT +      half * 8];
        v8bf a1l1 = *(const v8bf*)&Al[(mrow + 16) * LDT + 16 + half * 8];
        v8bf bh0  = *(const v8bf*)&Bh[nb * LDT + half * 8];
        v8bf bh1  = *(const v8bf*)&Bh[nb * LDT + 16 + half * 8];
        v8bf bl0  = *(const v8bf*)&Bl[nb * LDT + half * 8];
        v8bf bl1  = *(const v8bf*)&Bl[nb * LDT + 16 + half * 8];
        v16bf a0_hi = __builtin_shufflevector(a0h0, a0h1, 0,1,2,3,4,5,6,7,8,9,10,11,12,13,14,15);
        v16bf a0_lo = __builtin_shufflevector(a0l0, a0l1, 0,1,2,3,4,5,6,7,8,9,10,11,12,13,14,15);
        v16bf a1_hi = __builtin_shufflevector(a1h0, a1h1, 0,1,2,3,4,5,6,7,8,9,10,11,12,13,14,15);
        v16bf a1_lo = __builtin_shufflevector(a1l0, a1l1, 0,1,2,3,4,5,6,7,8,9,10,11,12,13,14,15);
        v16bf b_hi  = __builtin_shufflevector(bh0,  bh1,  0,1,2,3,4,5,6,7,8,9,10,11,12,13,14,15);
        v16bf b_lo  = __builtin_shufflevector(bl0,  bl1,  0,1,2,3,4,5,6,7,8,9,10,11,12,13,14,15);
        // split-bf16 fp32 emulation, two independent accumulator chains
        acc0 = __builtin_amdgcn_wmma_f32_16x16x32_bf16(false, a0_hi, false, b_hi, (short)0, acc0, false, false);
        acc1 = __builtin_amdgcn_wmma_f32_16x16x32_bf16(false, a1_hi, false, b_hi, (short)0, acc1, false, false);
        acc0 = __builtin_amdgcn_wmma_f32_16x16x32_bf16(false, a0_hi, false, b_lo, (short)0, acc0, false, false);
        acc1 = __builtin_amdgcn_wmma_f32_16x16x32_bf16(false, a1_hi, false, b_lo, (short)0, acc1, false, false);
        acc0 = __builtin_amdgcn_wmma_f32_16x16x32_bf16(false, a0_lo, false, b_hi, (short)0, acc0, false, false);
        acc1 = __builtin_amdgcn_wmma_f32_16x16x32_bf16(false, a1_lo, false, b_hi, (short)0, acc1, false, false);
    };

    // ---- software pipeline: async copy of chunk i+1 overlaps WMMA of chunk i ----
    issue(0, 0);
    wait_async0();
    __syncthreads();
    for (int kc = 0; kc < DK; kc += 2 * KT) {
        if (kc + KT < DK) issue(kc + KT, 1);
        compute(0);
        wait_async0();
        __syncthreads();
        if (kc + 2 * KT < DK) issue(kc + 2 * KT, 0);
        compute(1);
        wait_async0();
        __syncthreads();
    }

    // ---- distance + fused min/argmin ----
    const float cs = colSum[n0 + nb];
    const float cq = colSq[n0 + nb];
    const unsigned long long gcol = (unsigned long long)(unsigned int)(n0 + nb);
    #pragma unroll
    for (int t2 = 0; t2 < 2; ++t2) {
        v8f a = t2 ? acc1 : acc0;
        #pragma unroll
        for (int v = 0; v < 8; ++v) {
            const int r = v + 8 * half + 16 * t2;
            float sq = rowQ[r] + cq - 2.0f * a[v]
                     + 2.0f * EPSV * (rowS[r] - cs)
                     + (float)DK * EPSV * EPSV;
            float dist = sqrtf(fmaxf(sq, 0.0f));
            unsigned long long p = ((unsigned long long)__float_as_uint(dist) << 32) | gcol;
            #pragma unroll
            for (int off = 8; off > 0; off >>= 1) {
                unsigned long long o = __shfl_xor(p, off, 32);
                if (o < p) p = o;
            }
            if (mrow == 0) atomicMin(&rowMin[r], p);   // ds_min_u64
        }
    }
    __syncthreads();
    if (tid < 32) atomicMin(&gmin[m0 + tid], rowMin[tid]);  // global_atomic_min_u64
}

// ---- fallback (small workspace): fused inline-conversion kernel, 16-row tiles ----
__global__ __launch_bounds__(256)
void som_wmma_inline_kernel(const float* __restrict__ x, const float* __restrict__ w,
                            unsigned long long* __restrict__ gmin) {
    __shared__ alignas(16) unsigned short Ahi[16 * LDT];
    __shared__ alignas(16) unsigned short Alo[16 * LDT];
    __shared__ alignas(16) unsigned short Bhi[TN * LDT];
    __shared__ alignas(16) unsigned short Blo[TN * LDT];
    __shared__ float rowSum[16], rowSq[16], colSum[TN], colSq[TN];
    __shared__ unsigned long long rowMin[16];

    const int tid = threadIdx.x;
    const int m0  = blockIdx.x * 16;
    const int n0  = blockIdx.y * TN;

    if (tid < 16) { rowSum[tid] = 0.f; rowSq[tid] = 0.f; rowMin[tid] = ~0ull; }
    if (tid < TN) { colSum[tid] = 0.f; colSq[tid] = 0.f; }

    const int wn4 = tid & 31;
    const int wkb = tid >> 5;
    float cS[4] = {0.f,0.f,0.f,0.f}, cQ[4] = {0.f,0.f,0.f,0.f};

    const int lane = tid & 31, wave = tid >> 5;
    const int mrow = lane & 15, half = lane >> 4;
    v8f acc = {};

    for (int kc = 0; kc < DK; kc += KT) {
        __syncthreads();
        if (tid < 128) {
            const int m = tid >> 3, k4 = (tid & 7) << 2;
            const float4 xv = *(const float4*)(x + (size_t)(m0 + m) * DK + kc + k4);
            atomicAdd(&rowSum[m], xv.x + xv.y + xv.z + xv.w);
            atomicAdd(&rowSq[m],  xv.x*xv.x + xv.y*xv.y + xv.z*xv.z + xv.w*xv.w);
            float e[4] = {xv.x, xv.y, xv.z, xv.w};
            #pragma unroll
            for (int j = 0; j < 4; ++j) {
                unsigned short h = f2bf_rne(e[j]);
                float hf = __uint_as_float((unsigned int)h << 16);
                Ahi[m * LDT + k4 + j] = h;
                Alo[m * LDT + k4 + j] = f2bf_rne(e[j] - hf);
            }
        }
        #pragma unroll
        for (int s = 0; s < 4; ++s) {
            const int k = wkb + 8 * s;
            const float4 wv = *(const float4*)(w + (size_t)(kc + k) * MCOLS + n0 + 4 * wn4);
            float e[4] = {wv.x, wv.y, wv.z, wv.w};
            #pragma unroll
            for (int j = 0; j < 4; ++j) {
                const int n = 4 * wn4 + j;
                cS[j] += e[j]; cQ[j] += e[j] * e[j];
                unsigned short h = f2bf_rne(e[j]);
                float hf = __uint_as_float((unsigned int)h << 16);
                Bhi[n * LDT + k] = h;
                Blo[n * LDT + k] = f2bf_rne(e[j] - hf);
            }
        }
        __syncthreads();
        v8bf ah0 = *(const v8bf*)&Ahi[mrow * LDT +      half * 8];
        v8bf ah1 = *(const v8bf*)&Ahi[mrow * LDT + 16 + half * 8];
        v8bf al0 = *(const v8bf*)&Alo[mrow * LDT +      half * 8];
        v8bf al1 = *(const v8bf*)&Alo[mrow * LDT + 16 + half * 8];
        const int nbf = wave * 16 + mrow;
        v8bf bh0 = *(const v8bf*)&Bhi[nbf * LDT + half * 8];
        v8bf bh1 = *(const v8bf*)&Bhi[nbf * LDT + 16 + half * 8];
        v8bf bl0 = *(const v8bf*)&Blo[nbf * LDT + half * 8];
        v8bf bl1 = *(const v8bf*)&Blo[nbf * LDT + 16 + half * 8];
        v16bf a_hi = __builtin_shufflevector(ah0, ah1, 0,1,2,3,4,5,6,7,8,9,10,11,12,13,14,15);
        v16bf a_lo = __builtin_shufflevector(al0, al1, 0,1,2,3,4,5,6,7,8,9,10,11,12,13,14,15);
        v16bf b_hi = __builtin_shufflevector(bh0, bh1, 0,1,2,3,4,5,6,7,8,9,10,11,12,13,14,15);
        v16bf b_lo = __builtin_shufflevector(bl0, bl1, 0,1,2,3,4,5,6,7,8,9,10,11,12,13,14,15);
        acc = __builtin_amdgcn_wmma_f32_16x16x32_bf16(false, a_hi, false, b_hi, (short)0, acc, false, false);
        acc = __builtin_amdgcn_wmma_f32_16x16x32_bf16(false, a_hi, false, b_lo, (short)0, acc, false, false);
        acc = __builtin_amdgcn_wmma_f32_16x16x32_bf16(false, a_lo, false, b_hi, (short)0, acc, false, false);
    }
    #pragma unroll
    for (int j = 0; j < 4; ++j) {
        atomicAdd(&colSum[4 * wn4 + j], cS[j]);
        atomicAdd(&colSq[4 * wn4 + j],  cQ[j]);
    }
    __syncthreads();
    const int nblk = wave * 16 + mrow;
    const float cs = colSum[nblk], cq = colSq[nblk];
    const unsigned long long gcol = (unsigned long long)(unsigned int)(n0 + nblk);
    #pragma unroll
    for (int v = 0; v < 8; ++v) {
        const int r = v + 8 * half;
        float sq = rowSq[r] + cq - 2.0f * acc[v]
                 + 2.0f * EPSV * (rowSum[r] - cs) + (float)DK * EPSV * EPSV;
        float dist = sqrtf(fmaxf(sq, 0.0f));
        unsigned long long p = ((unsigned long long)__float_as_uint(dist) << 32) | gcol;
        #pragma unroll
        for (int off = 8; off > 0; off >>= 1) {
            unsigned long long o = __shfl_xor(p, off, 32);
            if (o < p) p = o;
        }
        if (mrow == 0) atomicMin(&rowMin[r], p);
    }
    __syncthreads();
    if (tid < 16) atomicMin(&gmin[m0 + tid], rowMin[tid]);
}

__global__ __launch_bounds__(256)
void som_finalize_kernel(const unsigned long long* __restrict__ gmin,
                         const float* __restrict__ loc, float* __restrict__ out) {
    int b = blockIdx.x * blockDim.x + threadIdx.x;
    if (b >= BROWS) return;
    unsigned long long p = gmin[b];
    unsigned int idx = (unsigned int)(p & 0xFFFFFFFFu);
    float dist = __uint_as_float((unsigned int)(p >> 32));
    out[b] = (float)idx;                           // bmu_indexes
    out[BROWS + 2 * b]     = loc[2 * idx];         // bmu_locations
    out[BROWS + 2 * b + 1] = loc[2 * idx + 1];
    atomicAdd(&out[3 * BROWS], dist * (1.0f / (float)BROWS));  // mean min-distance
}

extern "C" void kernel_launch(void* const* d_in, const int* in_sizes, int n_in,
                              void* d_out, int out_size, void* d_ws, size_t ws_size,
                              hipStream_t stream) {
    const float* x   = (const float*)d_in[0];
    const float* w   = (const float*)d_in[1];
    const float* loc = (const float*)d_in[2];
    float* out = (float*)d_out;
    char* ws = (char*)d_ws;
    unsigned long long* gmin = (unsigned long long*)(ws + OFF_GMIN);

    som_init_kernel<<<dim3(BROWS / 256), dim3(256), 0, stream>>>(gmin, out);

    if (ws_size >= WS_NEEDED) {
        unsigned short* whiT = (unsigned short*)(ws + OFF_WHI);
        unsigned short* wloT = (unsigned short*)(ws + OFF_WLO);
        unsigned short* xhi  = (unsigned short*)(ws + OFF_XHI);
        unsigned short* xlo  = (unsigned short*)(ws + OFF_XLO);
        float* rowSum = (float*)(ws + OFF_ROWSUM);
        float* rowSq  = (float*)(ws + OFF_ROWSQ);
        float* colSum = (float*)(ws + OFF_COLSUM);
        float* colSq  = (float*)(ws + OFF_COLSQ);

        som_prep_w<<<dim3(MCOLS / 256), dim3(256), 0, stream>>>(w, whiT, wloT, colSum, colSq);
        som_prep_x<<<dim3(BROWS), dim3(256), 0, stream>>>(x, xhi, xlo, rowSum, rowSq);
        dim3 grid(BROWS / 32, MCOLS / TN);   // 64 x 128 blocks
        som_gemm_kernel<<<grid, dim3(256), 0, stream>>>(xhi, xlo, whiT, wloT,
                                                        rowSum, rowSq, colSum, colSq, gmin);
    } else {
        dim3 grid(BROWS / 16, MCOLS / TN);   // 128 x 128 blocks
        som_wmma_inline_kernel<<<grid, dim3(256), 0, stream>>>(x, w, gmin);
    }

    som_finalize_kernel<<<dim3(BROWS / 256), dim3(256), 0, stream>>>(gmin, loc, out);
}
